// SelfAttention_8744553414863
// MI455X (gfx1250) — matrix operand order repaired and checked by reading, not compiled, not run on hardware
//
#include <hip/hip_runtime.h>
#include <hip/hip_bf16.h>

// ---------------------------------------------------------------------------
// MI455X (gfx1250) self-attention, flash-style, bf16 WMMA:
//   Q = x@W + b ; S = Q@Q^T ; A = softmax(S) ; out = A@x
// B=8, S=2048, D=512, fp32 in/out.
// All fragment operands are laid out so each lane's 16 elements are
// two contiguous 8-element runs -> 2x b128 loads per fragment (no u16 gathers).
// K tiles reach LDS via global_load_async_to_lds_b128 (ASYNCcnt path).
// ---------------------------------------------------------------------------

typedef __bf16 bf16_t;
typedef __attribute__((ext_vector_type(16))) __bf16 v16bf;
typedef __attribute__((ext_vector_type(8)))  float  v8f;

#define DIM 512
#define SEQ 2048
#define NB  8

union frag16 {
  v16bf v;
  uint4 q[2];
};

__device__ __forceinline__ v8f wmma_bf16(v16bf a, v16bf b, v8f c) {
  return __builtin_amdgcn_wmma_f32_16x16x32_bf16(false, a, false, b,
                                                 (short)0, c, false, false);
}

// A fragment 16x32 (MxK) from row-major source (ld in elements).
// ISA layout: lane half -> +8 K offset; element i -> k=(i>>3)*16+(i&7)+8*half.
// Per lane: two contiguous 8-element runs -> two 16B loads.
__device__ __forceinline__ v16bf load_a_frag(const bf16_t* base, int ld) {
  const int lane = threadIdx.x & 31;
  const int m = lane & 15, half = lane >> 4;
  frag16 f;
  const bf16_t* p = base + (size_t)m * ld + half * 8;
  f.q[0] = *(const uint4*)(p);
  f.q[1] = *(const uint4*)(p + 16);
  return f.v;
}

// B fragment 32x16 (KxN) where the SOURCE is stored N-major: elem(k,n) =
// base[n*ld + k]. Lane layout: n=lane&15, k=(i>>3)*16+(i&7)+8*half -> two
// contiguous 8-element runs -> two 16B loads.
__device__ __forceinline__ v16bf load_bn_frag(const bf16_t* base, int ld) {
  const int lane = threadIdx.x & 31;
  const int n = lane & 15, half = lane >> 4;
  frag16 f;
  const bf16_t* p = base + (size_t)n * ld + half * 8;
  f.q[0] = *(const uint4*)(p);
  f.q[1] = *(const uint4*)(p + 16);
  return f.v;
}

// gfx1250 async global->LDS copy (16B per lane), tracked by ASYNCcnt.
__device__ __forceinline__ void async_load_b128(bf16_t* lds_dst, const bf16_t* gsrc) {
  unsigned int lds_off = (unsigned int)(uintptr_t)lds_dst;  // low 32b = LDS offset
  asm volatile("global_load_async_to_lds_b128 %0, %1, off"
               :: "v"(lds_off), "v"(gsrc)
               : "memory");
}
__device__ __forceinline__ void wait_async0() {
  asm volatile("s_wait_asynccnt 0" ::: "memory");
}

// ---------------------------------------------------------------------------
// Kernel 1: build bf16 operands in workspace:
//   xbf  [B,S,D] row-major   (A operand of qproj)
//   xbfT [B,D,S]             (B operand of P@V, N-major)
//   WbfT [D(out),D(in)]      (B operand of qproj, N-major = W transposed)
// ---------------------------------------------------------------------------
__global__ __launch_bounds__(256) void convert_kernel(const float* __restrict__ x,
                                                      const float* __restrict__ W,
                                                      bf16_t* __restrict__ xbf,
                                                      bf16_t* __restrict__ xbfT,
                                                      bf16_t* __restrict__ WbfT,
                                                      int nx, int nw) {
  const int tid = blockIdx.x * blockDim.x + threadIdx.x;
  const int stride = gridDim.x * blockDim.x;
  for (int i = tid; i < nx; i += stride) {
    const float v = x[i];
    xbf[i] = (bf16_t)v;
    const int b = i / (SEQ * DIM);
    const int r = i - b * (SEQ * DIM);
    const int s = r / DIM;
    const int n = r - s * DIM;
    xbfT[((size_t)b * DIM + n) * SEQ + s] = (bf16_t)v;
  }
  for (int i = tid; i < nw; i += stride) {
    const int k = i / DIM;       // input dim
    const int n = i - k * DIM;   // output dim
    WbfT[(size_t)n * DIM + k] = (bf16_t)W[i];
  }
}

// ---------------------------------------------------------------------------
// Kernel 2: Q = x@W + b, bf16 out. One 16x16 C tile per wave, 16 chained
// WMMAs over K=512. Both operands load as 2x b128 per fragment.
// ---------------------------------------------------------------------------
__global__ __launch_bounds__(256) void qproj_kernel(const bf16_t* __restrict__ xbf,
                                                    const bf16_t* __restrict__ WbfT,
                                                    const float* __restrict__ bias,
                                                    bf16_t* __restrict__ Qbf) {
  const int wave = threadIdx.x >> 5;
  const int lane = threadIdx.x & 31;
  const int tile = blockIdx.x * 8 + wave;
  const int tm = tile >> 5;   // row tile of [16384, 512]
  const int tn = tile & 31;   // col tile

  v8f acc = {};
#pragma unroll
  for (int kk = 0; kk < DIM / 32; ++kk) {
    v16bf a = load_a_frag(xbf + (size_t)(tm * 16) * DIM + kk * 32, DIM);
    v16bf b = load_bn_frag(WbfT + (size_t)(tn * 16) * DIM + kk * 32, DIM);
    acc = wmma_bf16(a, b, acc);
  }

  const int n = lane & 15;
  const int rowbase = (lane >> 4) * 8;
  const float bv = bias[tn * 16 + n];
#pragma unroll
  for (int jv = 0; jv < 8; ++jv) {
    const int m = rowbase + jv;
    Qbf[(size_t)(tm * 16 + m) * DIM + tn * 16 + n] = (bf16_t)(acc[jv] + bv);
  }
}

// ---------------------------------------------------------------------------
// Kernel 3: flash attention. Block = 8 waves, 32 query rows.
// Loop over 32-key tiles:
//   (1) async-copy K tile (32x512 bf16, 32KB) into LDS; waves 4-7 prefetch
//       the next tile into L2 meanwhile.
//   (2) waves 0-3 compute the four 16x16 score fragments (16 WMMAs each,
//       B operand = ds_load_b128 pairs from the K tile).
//   (3) fp32 online softmax, one lane per row; P -> bf16 in LDS.
//   (4) all 8 waves rescale their 16x128 fp32 accumulators and do P@V;
//       V fragments come straight from L2-resident x^T (2x b128 per frag).
// ---------------------------------------------------------------------------
__global__ __launch_bounds__(256) void flash_kernel(const bf16_t* __restrict__ Qbf,
                                                    const bf16_t* __restrict__ xbfT,
                                                    float* __restrict__ out) {
  __shared__ __align__(16) bf16_t k_smem[32 * DIM];   // 32KB key tile
  __shared__ float  s_smem[32 * 32];
  __shared__ __align__(16) bf16_t p_smem[32 * 32];
  __shared__ float  alpha_smem[32];
  __shared__ float  m_smem[32];
  __shared__ float  l_smem[32];

  const int wave = threadIdx.x >> 5;
  const int lane = threadIdx.x & 31;
  const int batch = blockIdx.x >> 6;          // 512 blocks = 8 batches * 64
  const int row0  = (blockIdx.x & 63) * 32;

  const bf16_t* Qb  = Qbf  + (size_t)batch * SEQ * DIM;
  const bf16_t* XTb = xbfT + (size_t)batch * DIM * SEQ;

  if (threadIdx.x < 32) {
    m_smem[threadIdx.x] = -3.0e38f;
    l_smem[threadIdx.x] = 0.0f;
  }

  // Score waves 0-3: rowtile rt, key-half kh; resident Q A-fragments.
  const int rt = wave >> 1;
  const int kh = wave & 1;
  v16bf qa[DIM / 32];
  if (wave < 4) {
#pragma unroll
    for (int kk = 0; kk < DIM / 32; ++kk)
      qa[kk] = load_a_frag(Qb + (size_t)(row0 + rt * 16) * DIM + kk * 32, DIM);
  }

  // Output ownership: rowtile rt2, col group cg (128 cols = 8 frags).
  const int rt2 = wave & 1;
  const int cg  = wave >> 1;
  v8f acc[8];
  {
    v8f z = {};
#pragma unroll
    for (int f = 0; f < 8; ++f) acc[f] = z;
  }

  const int rowbase = (lane >> 4) * 8;
  const int nn = lane & 15;

  for (int j = 0; j < SEQ / 32; ++j) {
    const int kbase = j * 32;
    __syncthreads();  // prior iteration's k_smem/p_smem reads complete

    // (1) async K-tile fill: 2048 x 16B, 8 ops per thread
    {
      const bf16_t* src = Qb + (size_t)kbase * DIM;
#pragma unroll
      for (int it = 0; it < 8; ++it) {
        const int idx = threadIdx.x + it * 256;   // 16B chunk index
        async_load_b128(k_smem + idx * 8, src + idx * 8);
      }
      wait_async0();
    }
    __syncthreads();

    // (2) scores: S = Q_tile @ K_tile^T (keys are rows -> already N-major)
    if (wave < 4) {
      v8f sc = {};
#pragma unroll
      for (int kk = 0; kk < DIM / 32; ++kk) {
        v16bf b = load_bn_frag(k_smem + (size_t)(kh * 16) * DIM + kk * 32, DIM);
        sc = wmma_bf16(qa[kk], b, sc);
      }
#pragma unroll
      for (int jv = 0; jv < 8; ++jv)
        s_smem[(rt * 16 + rowbase + jv) * 32 + kh * 16 + nn] = sc[jv];
    } else if (j + 1 < SEQ / 32) {
      // waves 4-7: pull next K tile toward L2 while scores run
      const char* nk = (const char*)(Qb + (size_t)(kbase + 32) * DIM);
      __builtin_prefetch(nk + (threadIdx.x - 128) * 256, 0, 1);
    }
    __syncthreads();

    // (3) online softmax: one lane per query row
    if (threadIdx.x < 32) {
      const int r = threadIdx.x;
      const float mo = m_smem[r];
      float mt = mo;
#pragma unroll
      for (int t = 0; t < 32; ++t) mt = fmaxf(mt, s_smem[r * 32 + t]);
      const float al = __expf(mo - mt);
      float ls = 0.0f;
#pragma unroll
      for (int t = 0; t < 32; ++t) {
        const float p = __expf(s_smem[r * 32 + t] - mt);
        ls += p;
        p_smem[r * 32 + t] = (bf16_t)p;
      }
      m_smem[r] = mt;
      l_smem[r] = l_smem[r] * al + ls;
      alpha_smem[r] = al;
    }
    __syncthreads();

    // (4) rescale + O += P @ V, V fragments direct from L2-resident x^T
    {
      v16bf pa = load_a_frag(p_smem + rt2 * 16 * 32, 32);
      float alv[8];
#pragma unroll
      for (int jv = 0; jv < 8; ++jv)
        alv[jv] = alpha_smem[rt2 * 16 + rowbase + jv];
#pragma unroll
      for (int f = 0; f < 8; ++f) {
#pragma unroll
        for (int jv = 0; jv < 8; ++jv) acc[f][jv] *= alv[jv];
        v16bf vb = load_bn_frag(XTb + (size_t)(cg * 128 + f * 16) * SEQ + kbase, SEQ);
        acc[f] = wmma_bf16(pa, vb, acc[f]);
      }
    }
  }
  __syncthreads();

  // epilogue: divide by running sum, store fp32
  {
    float linv[8];
#pragma unroll
    for (int jv = 0; jv < 8; ++jv)
      linv[jv] = 1.0f / l_smem[rt2 * 16 + rowbase + jv];
#pragma unroll
    for (int f = 0; f < 8; ++f) {
#pragma unroll
      for (int jv = 0; jv < 8; ++jv) {
        const int row = row0 + rt2 * 16 + rowbase + jv;
        const int col = cg * 128 + f * 16 + nn;
        out[((size_t)batch * SEQ + row) * DIM + col] = acc[f][jv] * linv[jv];
      }
    }
  }
}

// ---------------------------------------------------------------------------
// Launch. ws layout: [xbf 16MB | xbfT 16MB | Qbf 16MB | WbfT 512KB]
// ---------------------------------------------------------------------------
extern "C" void kernel_launch(void* const* d_in, const int* in_sizes, int n_in,
                              void* d_out, int out_size, void* d_ws, size_t ws_size,
                              hipStream_t stream) {
  const float* x = (const float*)d_in[0];
  const float* W = (const float*)d_in[1];
  const float* b = (const float*)d_in[2];
  float* out = (float*)d_out;

  char* ws = (char*)d_ws;
  bf16_t* xbf  = (bf16_t*)ws;
  bf16_t* xbfT = (bf16_t*)(ws + (size_t)16 * 1024 * 1024);
  bf16_t* Qbf  = (bf16_t*)(ws + (size_t)32 * 1024 * 1024);
  bf16_t* WbfT = (bf16_t*)(ws + (size_t)48 * 1024 * 1024);

  const int nx = NB * SEQ * DIM;   // 8,388,608
  const int nw = DIM * DIM;        // 262,144

  convert_kernel<<<2048, 256, 0, stream>>>(x, W, xbf, xbfT, WbfT, nx, nw);
  qproj_kernel<<<4096, 256, 0, stream>>>(xbf, WbfT, b, Qbf);
  flash_kernel<<<512, 256, 0, stream>>>(Qbf, xbfT, out);
}
